// Model_42932493091118
// MI455X (gfx1250) — matrix operand order repaired and checked by reading, not compiled, not run on hardware
//
#include <hip/hip_runtime.h>
#include <hip/hip_bf16.h>

// ---------------------------------------------------------------------------
// Fused 2-layer MoE for MI455X (gfx1250, wave32).
//  * Expert GEMMs on v_wmma_f32_16x16x32_bf16, both FFN GEMMs fused via LDS
//    (the (B,E,1024) hidden tensor is never materialized).
//  * Weights pre-transposed to [out_col][k] bf16 so every LDS stage is a
//    contiguous 2-D strided tile -> staged by the Tensor Data Mover
//    (tensor_load_to_lds, TENSORcnt) with double-buffered LDS, overlapping
//    global->LDS DMA with WMMA compute. Manual b128 copy fallback if the
//    TDM builtin is unavailable.
// ---------------------------------------------------------------------------

#define B_TOK 8192
#define N_EXP 8
#define D_IN  1344
#define HDIM  1024
#define D_MID 128
#define D_OUTC 162

typedef __attribute__((ext_vector_type(16))) __bf16 v16bf;
typedef __attribute__((ext_vector_type(8)))  float  v8f;
typedef __attribute__((ext_vector_type(4))) unsigned int u32x4;
typedef __attribute__((ext_vector_type(8))) int i32x8;
typedef __attribute__((ext_vector_type(4))) int i32x4;

union FragAB { v16bf v; uint4 q[2]; };
union FragC  { v8f   v; float f[8]; };

// ---- TDM availability (guarded; falls back to manual b128 staging) --------
#if defined(__has_builtin)
#  if __has_builtin(__builtin_amdgcn_tensor_load_to_lds) && \
      __has_builtin(__builtin_amdgcn_s_wait_tensorcnt)
#    define HAVE_TDM 1
#  endif
#endif
#ifndef HAVE_TDM
#define HAVE_TDM 0
#endif
#if defined(__has_include)
#  if __has_include(<hip/amd_detail/amd_gfx1250_TDM.h>)
#    define TDM_ARGS6 1
#  endif
#endif
#ifndef TDM_ARGS6
#define TDM_ARGS6 0
#endif

__device__ __forceinline__ unsigned short f2bf(float f) {
  unsigned int u = __builtin_bit_cast(unsigned int, f);
  u += 0x7fffu + ((u >> 16) & 1u);           // round-to-nearest-even
  return (unsigned short)(u >> 16);
}
__device__ __forceinline__ float bf2f(unsigned short h) {
  unsigned int u = ((unsigned int)h) << 16;
  return __builtin_bit_cast(float, u);
}
__device__ __forceinline__ float gelu_erf(float x) {
  return 0.5f * x * (1.0f + erff(x * 0.70710678118654752440f));
}

#if HAVE_TDM
// Issue one TDM descriptor: 2-D bf16 tile (tile_k contiguous elems x rows),
// global row stride = row_stride elems, packed dense into LDS at lds_off.
// D# packing per CDNA5 ISA 8.3/8.4 (count=1, type=image, data_size=2B).
__device__ __forceinline__ void tdm_load_2d(unsigned int lds_off, const void* gptr,
                                            unsigned int tile_k, unsigned int rows,
                                            unsigned long long row_stride) {
  unsigned long long ga = (unsigned long long)gptr;
  u32x4 g0;
  g0[0] = 1u;                                           // count=1 (valid user D#)
  g0[1] = lds_off;                                      // lds_addr
  g0[2] = (unsigned int)(ga & 0xffffffffu);             // global_addr[31:0]
  g0[3] = (unsigned int)((ga >> 32) & 0x01ffffffu) | (2u << 30);  // addr[56:32]|type=2
  i32x8 g1;
  g1[0] = (int)(1u << 16);                              // wg_mask=0, data_size=1 (2B)
  g1[1] = (int)((tile_k & 0xffffu) << 16);              // tensor_dim0[15:0]
  g1[2] = (int)(((tile_k >> 16) & 0xffffu) | ((rows & 0xffffu) << 16));  // dim0 hi|dim1 lo
  g1[3] = (int)(((rows >> 16) & 0xffffu) | ((tile_k & 0xffffu) << 16));  // dim1 hi|tile_dim0
  g1[4] = (int)(rows & 0xffffu);                        // tile_dim1 (tile_dim2=0 -> 2D)
  g1[5] = (int)(row_stride & 0xffffffffu);              // tensor_dim0_stride lo
  g1[6] = (int)((row_stride >> 32) & 0xffffu);          // dim0_stride hi (dim1_stride=0)
  g1[7] = 0;
  i32x4 z4 = {0, 0, 0, 0};
#if TDM_ARGS6
  i32x8 z8 = {0, 0, 0, 0, 0, 0, 0, 0};
  __builtin_amdgcn_tensor_load_to_lds(g0, g1, z4, z4, z8, 0);
#else
  __builtin_amdgcn_tensor_load_to_lds(g0, g1, z4, z4, 0);
#endif
}
#endif

// ---------------- utility kernels -----------------------------------------

__global__ void k_zero(float* __restrict__ p, long n) {
  long i = (long)blockIdx.x * blockDim.x + threadIdx.x;
  if (i < n) p[i] = 0.0f;
}

// cast f32 -> bf16 with transpose: in (E, K, N) -> out (E, N, K)
__global__ void k_cast_tr(const float* __restrict__ in, unsigned short* __restrict__ out,
                          int K, int N) {
  long i = (long)blockIdx.x * blockDim.x + threadIdx.x;
  long total = (long)N_EXP * K * N;
  if (i >= total) return;
  int e = (int)(i / ((long)K * N));
  long r = i - (long)e * K * N;
  int k = (int)(r / N), n = (int)(r % N);
  out[(size_t)e * N * K + (size_t)n * K + k] = f2bf(in[i]);
}

__global__ void k_concat(const float* __restrict__ x8, const float* __restrict__ x16,
                         const float* __restrict__ x32, unsigned short* __restrict__ Xbf) {
  long i = (long)blockIdx.x * blockDim.x + threadIdx.x;
  if (i >= (long)B_TOK * D_IN) return;
  int t = (int)(i / D_IN);
  int j = (int)(i % D_IN);
  float v;
  if (j < 64)        v = x8[(size_t)t * 64 + j];
  else if (j < 320)  v = x16[(size_t)t * 256 + (j - 64)];
  else               v = x32[(size_t)t * 1024 + (j - 320)];
  Xbf[i] = f2bf(v);
}

// gate1: softmax(x @ gate1_w + b) over 8 experts, one thread per token
__global__ void k_gate1(const unsigned short* __restrict__ Xbf, const float* __restrict__ GW,
                        const float* __restrict__ GB, float* __restrict__ P) {
  int t = blockIdx.x * blockDim.x + threadIdx.x;
  if (t >= B_TOK) return;
  float s[N_EXP];
#pragma unroll
  for (int e = 0; e < N_EXP; ++e) s[e] = GB[e];
  const unsigned short* xr = Xbf + (size_t)t * D_IN;
  for (int d = 0; d < D_IN; ++d) {
    float xv = bf2f(xr[d]);
#pragma unroll
    for (int e = 0; e < N_EXP; ++e) s[e] += xv * GW[d * N_EXP + e];
  }
  float mx = s[0];
#pragma unroll
  for (int e = 1; e < N_EXP; ++e) mx = fmaxf(mx, s[e]);
  float sum = 0.0f;
#pragma unroll
  for (int e = 0; e < N_EXP; ++e) { s[e] = expf(s[e] - mx); sum += s[e]; }
  float inv = 1.0f / sum;
#pragma unroll
  for (int e = 0; e < N_EXP; ++e) P[(size_t)t * N_EXP + e] = s[e] * inv;
}

// post-combine GELU on y, plus bf16 copy for layer-2 WMMA
__global__ void k_gelu_y(float* __restrict__ Y, unsigned short* __restrict__ Ybf) {
  long i = (long)blockIdx.x * blockDim.x + threadIdx.x;
  if (i >= (long)B_TOK * D_MID) return;
  float v = gelu_erf(Y[i]);
  Y[i] = v;
  Ybf[i] = f2bf(v);
}

// gate2: softmax -> top-2 -> normalized weights (zero elsewhere) + aux sums
__global__ void k_gate2(const float* __restrict__ Y, const float* __restrict__ GW,
                        float* __restrict__ Wfull, float* __restrict__ Aux) {
  int t = blockIdx.x * blockDim.x + threadIdx.x;
  if (t >= B_TOK) return;
  float s[N_EXP];
#pragma unroll
  for (int e = 0; e < N_EXP; ++e) s[e] = 0.0f;
  const float* yr = Y + (size_t)t * D_MID;
  for (int d = 0; d < D_MID; ++d) {
    float yv = yr[d];
#pragma unroll
    for (int e = 0; e < N_EXP; ++e) s[e] += yv * GW[d * N_EXP + e];
  }
  float mx = s[0];
#pragma unroll
  for (int e = 1; e < N_EXP; ++e) mx = fmaxf(mx, s[e]);
  float p[N_EXP]; float sum = 0.0f;
#pragma unroll
  for (int e = 0; e < N_EXP; ++e) { p[e] = expf(s[e] - mx); sum += p[e]; }
  float inv = 1.0f / sum;
#pragma unroll
  for (int e = 0; e < N_EXP; ++e) p[e] *= inv;
  int i1 = 0;
#pragma unroll
  for (int e = 1; e < N_EXP; ++e) if (p[e] > p[i1]) i1 = e;   // lower index wins ties
  int i2 = (i1 == 0) ? 1 : 0;
#pragma unroll
  for (int e = 0; e < N_EXP; ++e) if (e != i1 && p[e] > p[i2]) i2 = e;
  float ssum = p[i1] + p[i2] + 1e-6f;
#pragma unroll
  for (int e = 0; e < N_EXP; ++e) Wfull[(size_t)t * N_EXP + e] = 0.0f;
  Wfull[(size_t)t * N_EXP + i1] = p[i1] / ssum;
  Wfull[(size_t)t * N_EXP + i2] = p[i2] / ssum;
#pragma unroll
  for (int e = 0; e < N_EXP; ++e) atomicAdd(&Aux[e], p[e]);
  atomicAdd(&Aux[N_EXP + i1], 1.0f);
  atomicAdd(&Aux[N_EXP + i2], 1.0f);
}

__global__ void k_aux_final(const float* __restrict__ Aux, float* __restrict__ out) {
  if (threadIdx.x == 0 && blockIdx.x == 0) {
    float a = 0.0f;
    for (int e = 0; e < N_EXP; ++e)
      a += (Aux[e] / (float)B_TOK) * (Aux[N_EXP + e] / (float)(B_TOK * 2));
    out[0] = (float)N_EXP * a;
  }
}

// ---------------- fused expert FFN (WMMA bf16, TDM-staged) ----------------
// One block = 128-token tile x one expert, 8 waves, each wave owns a 16-row
// strip. Double-buffered LDS K-step tiles; wave 0 issues tensor_load_to_lds
// for step ks+1 while all waves run WMMA on step ks.
// WT1: [E][HDIM][kdim]  (W1 transposed), WT2: [E][Nreal][HDIM] (W2 transposed)
template <int NT2>
__launch_bounds__(256)
__global__ void k_moe_ffn(const unsigned short* __restrict__ X, int kdim, int kSteps,
                          const unsigned short* __restrict__ WT1, const float* __restrict__ B1,
                          const unsigned short* __restrict__ WT2, const float* __restrict__ B2,
                          const float* __restrict__ tokW, float* __restrict__ Out, int Nreal) {
  constexpr int NCOLS2 = NT2 * 16;
  constexpr int WCOLS  = (NCOLS2 > 128) ? NCOLS2 : 128;

  __shared__ __align__(16) unsigned short Xs[2][128 * 32];    // A tiles (double buf)
  __shared__ __align__(16) unsigned short Ws[2][WCOLS * 32];  // B tiles [n][k] (double buf)
  __shared__ __align__(16) unsigned short Hb[128 * 128];      // hidden chunk (bf16)

  const int tid = threadIdx.x, lane = tid & 31, wv = tid >> 5;
  const int rowBase = blockIdx.x * 128;
  const int e = blockIdx.y;

  const unsigned short* WT1e = WT1 + (size_t)e * HDIM * kdim;
  const unsigned short* WT2e = WT2 + (size_t)e * Nreal * HDIM;
  const float* B1e = B1 + (size_t)e * HDIM;
  const float* B2e = B2 + (size_t)e * Nreal;

  const int r0 = wv * 16;
  const int m  = lane & 15;
  const int hi = lane >> 4;
  const int ka = hi ? 8 : 0;    // A-frag K base (ISA 16-bit A layout)
  const int kb = hi ? 8 : 0;    // B-frag K base (ISA 16-bit B layout)

  // zero both Ws buffers once (provides zero padding rows n >= Nreal)
  for (int idx = tid; idx < WCOLS * 8; idx += 256)
    ((uint4*)Ws)[idx] = make_uint4(0u, 0u, 0u, 0u);
  __syncthreads();

  // ---- staging helpers (TDM issue by wave 0, or vectorized b128 copies) ---
  auto stage1 = [&](int buf, int hc, int k0) {
#if HAVE_TDM
    if (wv == 0) {
      tdm_load_2d((unsigned int)(unsigned long long)&Xs[buf][0],
                  X + (size_t)rowBase * kdim + k0, 32u, 128u, (unsigned long long)kdim);
      tdm_load_2d((unsigned int)(unsigned long long)&Ws[buf][0],
                  WT1e + (size_t)(hc * 128) * kdim + k0, 32u, 128u, (unsigned long long)kdim);
    }
#else
    for (int idx = tid; idx < 512; idx += 256) {
      int r = idx >> 2, c8 = (idx & 3) << 3;
      *(uint4*)&Xs[buf][r * 32 + c8] =
          *(const uint4*)&X[(size_t)(rowBase + r) * kdim + k0 + c8];
    }
    for (int idx = tid; idx < 512; idx += 256) {
      int r = idx >> 2, c8 = (idx & 3) << 3;
      *(uint4*)&Ws[buf][r * 32 + c8] =
          *(const uint4*)&WT1e[(size_t)(hc * 128 + r) * kdim + k0 + c8];
    }
#endif
  };
  auto stage2 = [&](int buf, int hc, int k2) {
#if HAVE_TDM
    if (wv == 0)
      tdm_load_2d((unsigned int)(unsigned long long)&Ws[buf][0],
                  WT2e + hc * 128 + k2, 32u, (unsigned int)Nreal,
                  (unsigned long long)HDIM);
#else
    for (int idx = tid; idx < Nreal * 4; idx += 256) {
      int r = idx >> 2, c8 = (idx & 3) << 3;
      *(uint4*)&Ws[buf][r * 32 + c8] =
          *(const uint4*)&WT2e[(size_t)r * HDIM + hc * 128 + k2 + c8];
    }
#endif
  };
#if HAVE_TDM
#define STAGE_WAIT() do { if (wv == 0) __builtin_amdgcn_s_wait_tensorcnt(0); } while (0)
#else
#define STAGE_WAIT() do { } while (0)
#endif

  FragC oacc[NT2];
#pragma unroll
  for (int i = 0; i < NT2; ++i)
#pragma unroll
    for (int j = 0; j < 8; ++j) oacc[i].f[j] = 0.0f;

  for (int hc = 0; hc < HDIM / 128; ++hc) {
    FragC hacc[8];
#pragma unroll
    for (int i = 0; i < 8; ++i)
#pragma unroll
      for (int j = 0; j < 8; ++j) hacc[i].f[j] = 0.0f;

    // ======== GEMM 1: h = X @ W1[:, hc*128 .. +128] ========
    stage1(0, hc, 0);
    STAGE_WAIT();
    __syncthreads();
    for (int ks = 0; ks < kSteps; ++ks) {
      const int cur = ks & 1;
      if (ks + 1 < kSteps) stage1(cur ^ 1, hc, (ks + 1) * 32);  // prefetch next
      // preload A + all B fragments, then WMMA burst
      FragAB a, bf[8];
      a.q[0] = *(const uint4*)&Xs[cur][(r0 + m) * 32 + ka];
      a.q[1] = *(const uint4*)&Xs[cur][(r0 + m) * 32 + ka + 16];
#pragma unroll
      for (int nt = 0; nt < 8; ++nt) {
        int n = nt * 16 + m;
        bf[nt].q[0] = *(const uint4*)&Ws[cur][n * 32 + kb];
        bf[nt].q[1] = *(const uint4*)&Ws[cur][n * 32 + kb + 16];
      }
#pragma unroll
      for (int nt = 0; nt < 8; ++nt)
        hacc[nt].v = __builtin_amdgcn_wmma_f32_16x16x32_bf16(
            false, a.v, false, bf[nt].v, (short)0, hacc[nt].v, false, false);
      STAGE_WAIT();
      __syncthreads();
    }

    // ---- bias + erf-GELU -> Hb (bf16); each wave writes only its strip ----
#pragma unroll
    for (int nt = 0; nt < 8; ++nt) {
      int col = nt * 16 + m;
      float bb = B1e[hc * 128 + col];
#pragma unroll
      for (int v = 0; v < 8; ++v) {
        int rr = r0 + v + (hi ? 8 : 0);
        Hb[rr * 128 + col] = f2bf(gelu_erf(hacc[nt].f[v] + bb));
      }
    }

    // ======== GEMM 2: out += h_chunk @ W2[hc*128 .. +128, :] ========
    stage2(0, hc, 0);
    STAGE_WAIT();
    __syncthreads();
    for (int k2s = 0; k2s < 4; ++k2s) {
      const int cur = k2s & 1;
      if (k2s + 1 < 4) stage2(cur ^ 1, hc, (k2s + 1) * 32);  // prefetch next
      const int k2 = k2s * 32;
      FragAB a, bf[NT2];
      a.q[0] = *(const uint4*)&Hb[(r0 + m) * 128 + k2 + ka];
      a.q[1] = *(const uint4*)&Hb[(r0 + m) * 128 + k2 + ka + 16];
#pragma unroll
      for (int nt = 0; nt < NT2; ++nt) {
        int n = nt * 16 + m;
        bf[nt].q[0] = *(const uint4*)&Ws[cur][n * 32 + kb];
        bf[nt].q[1] = *(const uint4*)&Ws[cur][n * 32 + kb + 16];
      }
#pragma unroll
      for (int nt = 0; nt < NT2; ++nt)
        oacc[nt].v = __builtin_amdgcn_wmma_f32_16x16x32_bf16(
            false, a.v, false, bf[nt].v, (short)0, oacc[nt].v, false, false);
      STAGE_WAIT();
      __syncthreads();
    }
  }

  // ---- gate-weighted combine: Out[row, n] += w(row,e) * (acc + b2[n]) ----
  float wrow[8];
#pragma unroll
  for (int v = 0; v < 8; ++v)
    wrow[v] = tokW[(size_t)(rowBase + r0 + v + (hi ? 8 : 0)) * N_EXP + e];
#pragma unroll
  for (int nt = 0; nt < NT2; ++nt) {
    int n = nt * 16 + m;
    if (n < Nreal) {
      float bb = B2e[n];
#pragma unroll
      for (int v = 0; v < 8; ++v) {
        int rr = rowBase + r0 + v + (hi ? 8 : 0);
        atomicAdd(&Out[(size_t)rr * Nreal + n], wrow[v] * (oacc[nt].f[v] + bb));
      }
    }
  }
#undef STAGE_WAIT
}

// ---------------- host launch ---------------------------------------------

extern "C" void kernel_launch(void* const* d_in, const int* in_sizes, int n_in,
                              void* d_out, int out_size, void* d_ws, size_t ws_size,
                              hipStream_t stream) {
  (void)in_sizes; (void)n_in; (void)out_size; (void)ws_size;

  const float* x8      = (const float*)d_in[0];
  const float* x16     = (const float*)d_in[1];
  const float* x32     = (const float*)d_in[2];
  const float* gate1_w = (const float*)d_in[3];
  const float* gate1_b = (const float*)d_in[4];
  const float* e1_w1   = (const float*)d_in[5];
  const float* e1_b1   = (const float*)d_in[6];
  const float* e1_w2   = (const float*)d_in[7];
  const float* e1_b2   = (const float*)d_in[8];
  const float* gate2_w = (const float*)d_in[9];
  const float* e2_w1   = (const float*)d_in[10];
  const float* e2_b1   = (const float*)d_in[11];
  const float* e2_w2   = (const float*)d_in[12];
  const float* e2_b2   = (const float*)d_in[13];
  float* out = (float*)d_out;

  // workspace carve-up (all chunks 256B-aligned)
  char* w = (char*)d_ws;
  unsigned short* xbf = (unsigned short*)w; w += (size_t)B_TOK * D_IN * 2;         // x (B,1344) bf16
  unsigned short* wt1 = (unsigned short*)w; w += (size_t)N_EXP * D_IN * HDIM * 2;  // [E][H][1344]
  unsigned short* wt2 = (unsigned short*)w; w += (size_t)N_EXP * HDIM * D_MID * 2; // [E][128][H]
  unsigned short* wt3 = (unsigned short*)w; w += (size_t)N_EXP * D_MID * HDIM * 2; // [E][H][128]
  unsigned short* wt4 = (unsigned short*)w; w += (size_t)N_EXP * HDIM * D_OUTC * 2;// [E][162][H]
  float* probs1 = (float*)w;                w += (size_t)B_TOK * N_EXP * 4;
  float* ypre   = (float*)w;                w += (size_t)B_TOK * D_MID * 4;
  unsigned short* ybf = (unsigned short*)w; w += (size_t)B_TOK * D_MID * 2;
  float* wfull  = (float*)w;                w += (size_t)B_TOK * N_EXP * 4;
  float* aux    = (float*)w;                w += 256;

  const int T = 256;
  auto blocks = [](long n) { return (unsigned)((n + 255) / 256); };

  // 1) bf16 x (concat) and transposed bf16 weights
  k_concat<<<blocks((long)B_TOK * D_IN), T, 0, stream>>>(x8, x16, x32, xbf);
  k_cast_tr<<<blocks((long)N_EXP * D_IN * HDIM), T, 0, stream>>>(e1_w1, wt1, D_IN, HDIM);
  k_cast_tr<<<blocks((long)N_EXP * HDIM * D_MID), T, 0, stream>>>(e1_w2, wt2, HDIM, D_MID);
  k_cast_tr<<<blocks((long)N_EXP * D_MID * HDIM), T, 0, stream>>>(e2_w1, wt3, D_MID, HDIM);
  k_cast_tr<<<blocks((long)N_EXP * HDIM * D_OUTC), T, 0, stream>>>(e2_w2, wt4, HDIM, D_OUTC);

  // 2) zero accumulation buffers (deterministic per launch)
  k_zero<<<blocks((long)B_TOK * D_MID), T, 0, stream>>>(ypre, (long)B_TOK * D_MID);
  k_zero<<<blocks((long)B_TOK * D_OUTC + 1), T, 0, stream>>>(out, (long)B_TOK * D_OUTC + 1);
  k_zero<<<1, 64, 0, stream>>>(aux, 64);

  // 3) layer-1 gate + fused dense MoE (atomic softmax-weighted combine)
  k_gate1<<<blocks(B_TOK), T, 0, stream>>>(xbf, gate1_w, gate1_b, probs1);
  k_moe_ffn<8><<<dim3(B_TOK / 128, N_EXP), T, 0, stream>>>(
      xbf, D_IN, D_IN / 32, wt1, e1_b1, wt2, e1_b2, probs1, ypre, D_MID);

  // 4) post-combine GELU, layer-2 router (top-2 + aux), fused top-k MoE
  k_gelu_y<<<blocks((long)B_TOK * D_MID), T, 0, stream>>>(ypre, ybf);
  k_gate2<<<blocks(B_TOK), T, 0, stream>>>(ypre, gate2_w, wfull, aux);
  k_moe_ffn<11><<<dim3(B_TOK / 128, N_EXP), T, 0, stream>>>(
      ybf, D_MID, D_MID / 32, wt3, e2_b1, wt4, e2_b2, wfull, out, D_OUTC);

  // 5) aux loss scalar
  k_aux_final<<<1, 32, 0, stream>>>(aux, out + (size_t)B_TOK * D_OUTC);
}
